// MCQTagger_60636348285044
// MI455X (gfx1250) — compile-verified, matrix operand order checked by reading, not run on hardware
//
#include <hip/hip_runtime.h>
#include <hip/hip_bf16.h>
#include <math.h>

typedef __attribute__((ext_vector_type(16))) _Float16 v16h;
typedef __attribute__((ext_vector_type(8)))  _Float16 v8h;
typedef __attribute__((ext_vector_type(8)))  float    v8f;

#define Bsz   32
#define Lsz   512
#define Csz   16
#define EMB   100
#define CEMB  30
#define COUT  30
#define BBOX  100
#define IN0   290
#define IN0P  320      // K padded to multiple of 32
#define HID   256
#define G4H   1024
#define IN1   512
#define NLAB  9
#define BL    (Bsz*Lsz)

// ---------------------------------------------------------------------------
// WMMA fragment loaders (CDNA5 16x16x32 f16 layouts, ISA 7.12.2)
// A (16x32, row-major source, leading-dim ld halfs):
//   lane<16: M=lane,  K = {0..7, 16..23};  lane>=16: M=lane-16, K = {8..15, 24..31}
// ---------------------------------------------------------------------------
__device__ __forceinline__ v16h ld_a_frag(const _Float16* base, int ld) {
  int lane = threadIdx.x & 31;
  int m  = lane & 15;
  int kb = (lane & 16) >> 1;              // 0 or 8
  const _Float16* p = base + m * ld + kb;
  v8h lo = *(const v8h*)(p);              // K = kb .. kb+7
  v8h hi = *(const v8h*)(p + 16);         // K = 16+kb .. 16+kb+7
  v16h r;
#pragma unroll
  for (int i = 0; i < 8; ++i) { r[i] = lo[i]; r[i + 8] = hi[i]; }
  return r;
}

// B fragments are pre-packed lane-major: frag*512 + lane*16 halfs, elems 0..15.
__device__ __forceinline__ v16h ld_b_frag(const _Float16* packed, int frag) {
  int lane = threadIdx.x & 31;
  const _Float16* p = packed + ((size_t)frag * 32 + lane) * 16;
  v8h lo = *(const v8h*)(p);
  v8h hi = *(const v8h*)(p + 8);
  v16h r;
#pragma unroll
  for (int i = 0; i < 8; ++i) { r[i] = lo[i]; r[i + 8] = hi[i]; }
  return r;
}

__device__ __forceinline__ float sigf(float x) { return 1.f / (1.f + __expf(-x)); }

// Async global->LDS copy (CDNA5, ASYNCcnt-tracked). lds_off is the byte offset
// inside this kernel's dynamic LDS allocation (base 0: kernel has no static LDS).
__device__ __forceinline__ void async_g2l_b128(unsigned lds_off, const void* g) {
  asm volatile("global_load_async_to_lds_b128 %0, %1, off"
               :: "v"(lds_off), "v"((unsigned long long)(size_t)g)
               : "memory");
}
__device__ __forceinline__ void wait_async0() {
  asm volatile("s_wait_asynccnt 0" ::: "memory");
}

// ---------------------------------------------------------------------------
// Pack W (row-major [N=1024][K=din] fp32, i.e. PyTorch W[gate_out][in]) into
// f16 WMMA B-fragment-major layout for W^T (K x N), K zero-padded to nkt*32.
// ---------------------------------------------------------------------------
__global__ void pack_wmma_b(const float* __restrict__ src, _Float16* __restrict__ dst,
                            int din, int nkt) {
  int idx = blockIdx.x * 256 + threadIdx.x;
  int total = nkt * 64 * 512;
  if (idx >= total) return;
  int e    = idx & 15;
  int lane = (idx >> 4) & 31;
  int frag = idx >> 9;                    // nt*nkt + kt
  int kt = frag % nkt, nt = frag / nkt;
  int K = kt * 32 + ((e >> 3) << 4) + ((lane & 16) >> 1) + (e & 7);
  int N = nt * 16 + (lane & 15);
  float v = (K < din) ? src[(size_t)N * din + K] : 0.f;
  dst[idx] = (_Float16)v;
}

// ---------------------------------------------------------------------------
// Embedding gather + char-CNN (k=3,4,5, relu+maxpool) + bbox linear -> f16 X
// One wave per token.
// ---------------------------------------------------------------------------
__global__ void embed_kernel(const int* __restrict__ words, const int* __restrict__ chars,
                             const float* __restrict__ bboxes,
                             const float* __restrict__ wemb, const float* __restrict__ cemb,
                             const float* __restrict__ cw0, const float* __restrict__ cw1,
                             const float* __restrict__ cw2,
                             const float* __restrict__ cb0, const float* __restrict__ cb1,
                             const float* __restrict__ cb2,
                             const float* __restrict__ bW, const float* __restrict__ bb,
                             _Float16* __restrict__ X) {
  __shared__ float ce[Csz][CEMB];
  int tok = blockIdx.x, lane = threadIdx.x;
  for (int i = lane; i < Csz * CEMB; i += 32) {
    int c = i / CEMB, e = i % CEMB;
    ce[c][e] = cemb[chars[tok * Csz + c] * CEMB + e];
  }
  __syncthreads();
  _Float16* xr = X + (size_t)tok * IN0P;
  int wid = words[tok];
  for (int j = lane; j < EMB; j += 32) xr[j] = (_Float16)wemb[wid * EMB + j];
  float b0 = bboxes[tok * 4 + 0], b1 = bboxes[tok * 4 + 1];
  float b2 = bboxes[tok * 4 + 2], b3 = bboxes[tok * 4 + 3];
  for (int j = lane; j < BBOX; j += 32)
    xr[EMB + 3 * COUT + j] =
        (_Float16)(b0 * bW[j * 4] + b1 * bW[j * 4 + 1] + b2 * bW[j * 4 + 2] +
                   b3 * bW[j * 4 + 3] + bb[j]);
  for (int j = IN0 + lane; j < IN0P; j += 32) xr[j] = (_Float16)0.f;  // K padding
  if (lane < COUT) {
    const float* ws[3] = {cw0, cw1, cw2};
    const float* bs[3] = {cb0, cb1, cb2};
#pragma unroll
    for (int ki = 0; ki < 3; ++ki) {
      int k = 3 + ki;
      float mraw = -1e30f;
      for (int p = 0; p + k <= Csz; ++p) {
        float acc = bs[ki][lane];
        for (int e = 0; e < CEMB; ++e)
          for (int i = 0; i < k; ++i)
            acc += ce[p + i][e] * ws[ki][(lane * CEMB + e) * k + i];
        mraw = fmaxf(mraw, acc);
      }
      xr[EMB + ki * COUT + lane] = (_Float16)fmaxf(mraw, 0.f);  // relu(max) == max(relu)
    }
  }
}

__global__ void lengths_kernel(const unsigned char* __restrict__ mask, int* __restrict__ len) {
  int b = threadIdx.x;
  if (b < Bsz) {
    int s = 0;
    for (int t = 0; t < Lsz; ++t) s += mask[b * Lsz + t] ? 1 : 0;
    len[b] = s;
  }
}

// dst[b][t][dOff..] = src[b][rev(t)][0..width)
__global__ void reverse_kernel(const _Float16* __restrict__ src, int sStride,
                               _Float16* __restrict__ dst, int dStride, int dOff,
                               int width, const int* __restrict__ len) {
  int b = blockIdx.x / Lsz, t = blockIdx.x % Lsz;
  int Lb = len[b];
  int ts = (t < Lb) ? (Lb - 1 - t) : t;
  const _Float16* s = src + ((size_t)b * Lsz + ts) * sStride;
  _Float16* d = dst + ((size_t)b * Lsz + t) * dStride + dOff;
  for (int j = threadIdx.x; j < width; j += blockDim.x) d[j] = s[j];
}

// ---------------------------------------------------------------------------
// Y[BL][1024] = X[BL][ldx] * W^T + (b1 + b2), f16 in / f32 acc / f16 out.
// 8 waves/block; each wave computes 1 M-tile x 4 N-tiles (A frag reused 4x).
// Grid = (BL/16)*(64/4)/8 = 2048.
// ---------------------------------------------------------------------------
__global__ void gemm_wmma(const _Float16* __restrict__ X, int ldx,
                          const _Float16* __restrict__ Wp, int nkt,
                          const float* __restrict__ b1, const float* __restrict__ b2,
                          _Float16* __restrict__ Y) {
  int wave = threadIdx.x >> 5;
  int g = blockIdx.x * 8 + wave;          // (mt, nt-group-of-4)
  int mt = g >> 4, ntg = g & 15;
  int lane = threadIdx.x & 31;
  v8f acc[4] = {};
  const _Float16* xb = X + (size_t)(mt * 16) * ldx;
  for (int kt = 0; kt < nkt; ++kt) {
    v16h a = ld_a_frag(xb + kt * 32, ldx);
#pragma unroll
    for (int j = 0; j < 4; ++j) {
      v16h bf = ld_b_frag(Wp, (ntg * 4 + j) * nkt + kt);
      acc[j] = __builtin_amdgcn_wmma_f32_16x16x32_f16(false, a, false, bf, (short)0,
                                                      acc[j], false, false);
    }
  }
  int rb = mt * 16 + ((lane & 16) >> 1);
#pragma unroll
  for (int j = 0; j < 4; ++j) {
    int n = (ntg * 4 + j) * 16 + (lane & 15);
    float bias = b1[n] + b2[n];
#pragma unroll
    for (int v = 0; v < 8; ++v)
      Y[(size_t)(rb + v) * G4H + n] = (_Float16)(acc[j][v] + bias);
  }
}

// ---------------------------------------------------------------------------
// LSTM recurrence, one workgroup (32 waves) per direction, grid = 2.
// Per step: gates[32][1024] = xw[:,t] + h @ Whh^T  (WMMA, 4 N-tiles/wave with
// hoisted A fragments), then pointwise gate math; h kept f16 in LDS.
// xw[:,t] is double-buffered in LDS via async global->LDS b128 copies issued
// one timestep ahead (ASYNCcnt), so the serial chain never waits on HBM/L2.
// Dynamic LDS map (kernel has no static LDS, so offsets are DS addresses):
//   [0,128K)    gates f32 [32][1024]
//   [128K,144K) h     f16 [32][256]
//   [144K,176K) c     f32 [32][256]
//   [176K,304K) xw    f16 [2][32][1024]   -> total 311296 B (< 320KB/WGP)
// ---------------------------------------------------------------------------
#define LSTM_GATES_OFF 0u
#define LSTM_H_OFF     131072u
#define LSTM_C_OFF     147456u
#define LSTM_XW_OFF    180224u
#define LSTM_XW_BUFB   65536u
#define LSTM_LDS_BYTES (LSTM_XW_OFF + 2 * LSTM_XW_BUFB)

__global__ void __launch_bounds__(1024) lstm_scan2(
    const _Float16* __restrict__ xwF, const _Float16* __restrict__ xwB,
    const _Float16* __restrict__ whF, const _Float16* __restrict__ whB,
    const unsigned char* __restrict__ mask,
    _Float16* __restrict__ outF, int offF, int strF,
    _Float16* __restrict__ outB, int offB, int strB) {
  extern __shared__ char smem[];
  float*    gates = (float*)(smem + LSTM_GATES_OFF);   // [32][1024]
  _Float16* hS    = (_Float16*)(smem + LSTM_H_OFF);    // [32][256]
  float*    cS    = (float*)(smem + LSTM_C_OFF);       // [32][256]
  const _Float16* XW  = blockIdx.x ? xwB : xwF;
  const _Float16* WHH = blockIdx.x ? whB : whF;
  _Float16* OUT = blockIdx.x ? outB : outF;
  int oOff = blockIdx.x ? offB : offF;
  int oStr = blockIdx.x ? strB : strF;
  int tid = threadIdx.x, wave = tid >> 5, lane = tid & 31;

  // stage xw[:,t] (32 rows x 2048B) into LDS buffer `buf`; coalesced b128s
  auto stage = [&](int t, int buf) {
    unsigned base = LSTM_XW_OFF + (unsigned)buf * LSTM_XW_BUFB;
#pragma unroll
    for (int k = 0; k < 4; ++k) {
      int c = tid + k * 1024;           // 4096 x 16B chunks
      int b = c >> 7;                   // 128 chunks per batch row
      int off = (c & 127) << 4;
      const char* g = (const char*)XW + ((size_t)b * Lsz + t) * (G4H * 2) + off;
      async_g2l_b128(base + (unsigned)(b * 2048 + off), g);
    }
  };

  for (int i = tid; i < Bsz * HID; i += 1024) { hS[i] = (_Float16)0.f; cS[i] = 0.f; }
  stage(0, 0);
  wait_async0();
  __syncthreads();

  int mt = wave >> 4;                              // waves 0..15 -> mt0, 16..31 -> mt1
  const _Float16* hb = hS + mt * 16 * HID;
  int rb = mt * 16 + ((lane & 16) >> 1);

  for (int t = 0; t < Lsz; ++t) {
    if (t + 1 < Lsz) stage(t + 1, (t + 1) & 1);    // overlap with this step's math
    const _Float16* xwCur =
        (const _Float16*)(smem + LSTM_XW_OFF + (unsigned)(t & 1) * LSTM_XW_BUFB);

    v16h afr[8];                                    // h A-fragments, shared by 4 tiles
#pragma unroll
    for (int kt = 0; kt < 8; ++kt) afr[kt] = ld_a_frag(hb + kt * 32, HID);

#pragma unroll
    for (int i = 0; i < 4; ++i) {
      int nt = (wave * 4 + i) & 63;
      int n  = nt * 16 + (lane & 15);
      v8f acc;
#pragma unroll
      for (int v = 0; v < 8; ++v)
        acc[v] = (float)xwCur[(rb + v) * G4H + n];
#pragma unroll
      for (int kt = 0; kt < 8; ++kt) {
        v16h bf = ld_b_frag(WHH, nt * 8 + kt);
        acc = __builtin_amdgcn_wmma_f32_16x16x32_f16(false, afr[kt], false, bf, (short)0,
                                                     acc, false, false);
      }
#pragma unroll
      for (int v = 0; v < 8; ++v)
        gates[(rb + v) * G4H + n] = acc[v];
    }
    __syncthreads();
#pragma unroll
    for (int i = 0; i < 8; ++i) {
      int id = tid + i * 1024;            // 32*256 (b, j) pairs
      int b = id >> 8, j = id & 255;
      float gi = gates[b * G4H + j];
      float gf = gates[b * G4H + HID + j];
      float gg = gates[b * G4H + 2 * HID + j];
      float go = gates[b * G4H + 3 * HID + j];
      float cp = cS[b * HID + j];
      float cn = sigf(gf) * cp + sigf(gi) * tanhf(gg);
      float hn = sigf(go) * tanhf(cn);
      bool  m  = mask[b * Lsz + t] != 0;
      float hp = (float)hS[b * HID + j];
      float hq = m ? hn : hp;
      cS[b * HID + j] = m ? cn : cp;
      hS[b * HID + j] = (_Float16)hq;
      OUT[((size_t)b * Lsz + t) * oStr + oOff + j] = (_Float16)(m ? hq : 0.f);
    }
    if (t + 1 < Lsz) wait_async0();       // next xw buffer resident before barrier
    __syncthreads();
  }
}

// emissions[BL][16(pad)] = H[BL][512] @ ffW^T + ffb  (9 labels, tiny)
__global__ void emis_kernel(const _Float16* __restrict__ H, const float* __restrict__ W,
                            const float* __restrict__ bias, float* __restrict__ em) {
  int row = blockIdx.x, lane = threadIdx.x;
  if (lane < NLAB) {
    float acc = bias[lane];
    const _Float16* h = H + (size_t)row * IN1;
    const float* w = W + lane * IN1;
    for (int k = 0; k < IN1; ++k) acc += (float)h[k] * w[k];
    em[(size_t)row * 16 + lane] = acc;
  }
}

__global__ void viterbi_kernel(const float* __restrict__ em,
                               const unsigned char* __restrict__ mask,
                               const float* __restrict__ trans,
                               const float* __restrict__ startv,
                               const float* __restrict__ endv,
                               int* __restrict__ out, const int* __restrict__ len) {
  __shared__ float tr[NLAB * NLAB];
  __shared__ float sc[NLAB], scn[NLAB];
  __shared__ unsigned char bp[Lsz][NLAB];
  int b = blockIdx.x, lane = threadIdx.x;
  for (int i = lane; i < NLAB * NLAB; i += 32) tr[i] = trans[i];
  if (lane < NLAB) sc[lane] = startv[lane] + em[((size_t)b * Lsz) * 16 + lane];
  __syncthreads();
  for (int t = 1; t < Lsz; ++t) {
    if (lane < NLAB) {
      float best = -1e30f; int bi = 0;
#pragma unroll
      for (int i = 0; i < NLAB; ++i) {
        float v = sc[i] + tr[i * NLAB + lane];
        if (v > best) { best = v; bi = i; }
      }
      best += em[((size_t)b * Lsz + t) * 16 + lane];
      bool m = mask[b * Lsz + t] != 0;
      scn[lane] = m ? best : sc[lane];
      bp[t][lane] = (unsigned char)bi;
    }
    __syncthreads();
    if (lane < NLAB) sc[lane] = scn[lane];
    __syncthreads();
  }
  if (lane == 0) {
    float best = -1e30f; int cur = 0;
    for (int i = 0; i < NLAB; ++i) {
      float v = sc[i] + endv[i];
      if (v > best) { best = v; cur = i; }
    }
    for (int t = Lsz - 1; t >= 1; --t) {
      out[b * Lsz + t] = mask[b * Lsz + t] ? cur : 0;
      int prev = bp[t][cur];
      if (mask[b * Lsz + t]) cur = prev;
    }
    out[b * Lsz] = mask[b * Lsz] ? cur : 0;
  }
}

// ---------------------------------------------------------------------------
extern "C" void kernel_launch(void* const* d_in, const int* in_sizes, int n_in,
                              void* d_out, int out_size, void* d_ws, size_t ws_size,
                              hipStream_t stream) {
  (void)in_sizes; (void)n_in; (void)out_size; (void)ws_size;
  const int*   words  = (const int*)d_in[0];
  const int*   chars  = (const int*)d_in[1];
  const float* bboxes = (const float*)d_in[2];
  const unsigned char* mask = (const unsigned char*)d_in[3];
  const float* wemb = (const float*)d_in[4];
  const float* cemb = (const float*)d_in[5];
  const float* cw[3] = {(const float*)d_in[6], (const float*)d_in[7], (const float*)d_in[8]};
  const float* cb[3] = {(const float*)d_in[9], (const float*)d_in[10], (const float*)d_in[11]};
  const float* bW = (const float*)d_in[12];
  const float* bb = (const float*)d_in[13];
  const float *Wih[4], *Whh[4], *bih[4], *bhh[4];
  for (int d = 0; d < 4; ++d) {
    Wih[d] = (const float*)d_in[14 + d * 4];
    Whh[d] = (const float*)d_in[15 + d * 4];
    bih[d] = (const float*)d_in[16 + d * 4];
    bhh[d] = (const float*)d_in[17 + d * 4];
  }
  const float* ffW    = (const float*)d_in[30];
  const float* ffb    = (const float*)d_in[31];
  const float* trans  = (const float*)d_in[32];
  const float* startv = (const float*)d_in[33];
  const float* endv   = (const float*)d_in[34];
  int* out = (int*)d_out;

  char* ws = (char*)d_ws;
  size_t off = 0;
  auto take = [&](size_t bytes) -> char* {
    char* p = ws + off;
    off = (off + bytes + 255) & ~(size_t)255;
    return p;
  };
  int nktIH[4] = {10, 10, 16, 16};  // Kpad/32: 320, 320, 512, 512
  _Float16* X     = (_Float16*)take((size_t)BL * IN0P * 2);
  _Float16* XR    = (_Float16*)take((size_t)BL * IN0P * 2);
  _Float16* WihP[4];
  _Float16* WhhP[4];
  for (int d = 0; d < 4; ++d) WihP[d] = (_Float16*)take((size_t)nktIH[d] * 64 * 512 * 2);
  for (int d = 0; d < 4; ++d) WhhP[d] = (_Float16*)take((size_t)8 * 64 * 512 * 2);
  _Float16* XWF    = (_Float16*)take((size_t)BL * G4H * 2);
  _Float16* XWB    = (_Float16*)take((size_t)BL * G4H * 2);
  _Float16* HCAT   = (_Float16*)take((size_t)BL * IN1 * 2);
  _Float16* HCATR  = (_Float16*)take((size_t)BL * IN1 * 2);
  _Float16* HBREV  = (_Float16*)take((size_t)BL * HID * 2);
  _Float16* H1CAT  = (_Float16*)take((size_t)BL * IN1 * 2);
  _Float16* H1BREV = (_Float16*)take((size_t)BL * HID * 2);
  float*    EM     = (float*)take((size_t)BL * 16 * 4);
  int*      LEN    = (int*)take(Bsz * 4);

  lengths_kernel<<<1, 32, 0, stream>>>(mask, LEN);
  for (int d = 0; d < 4; ++d) {
    int din = (d < 2) ? IN0 : IN1;
    int totI = nktIH[d] * 64 * 512;
    pack_wmma_b<<<(totI + 255) / 256, 256, 0, stream>>>(Wih[d], WihP[d], din, nktIH[d]);
    int totH = 8 * 64 * 512;
    pack_wmma_b<<<(totH + 255) / 256, 256, 0, stream>>>(Whh[d], WhhP[d], HID, 8);
  }
  embed_kernel<<<BL, 32, 0, stream>>>(words, chars, bboxes, wemb, cemb,
                                      cw[0], cw[1], cw[2], cb[0], cb[1], cb[2],
                                      bW, bb, X);
  reverse_kernel<<<BL, 64, 0, stream>>>(X, IN0P, XR, IN0P, 0, IN0P, LEN);

  // ---- layer 0 ----
  gemm_wmma<<<2048, 256, 0, stream>>>(X,  IN0P, WihP[0], 10, bih[0], bhh[0], XWF);
  gemm_wmma<<<2048, 256, 0, stream>>>(XR, IN0P, WihP[1], 10, bih[1], bhh[1], XWB);
  lstm_scan2<<<2, 1024, LSTM_LDS_BYTES, stream>>>(XWF, XWB, WhhP[0], WhhP[1], mask,
                                                  HCAT, 0, IN1, HBREV, 0, HID);
  reverse_kernel<<<BL, 64, 0, stream>>>(HBREV, HID, HCAT, IN1, HID, HID, LEN);
  reverse_kernel<<<BL, 64, 0, stream>>>(HCAT, IN1, HCATR, IN1, 0, IN1, LEN);

  // ---- layer 1 ----
  gemm_wmma<<<2048, 256, 0, stream>>>(HCAT,  IN1, WihP[2], 16, bih[2], bhh[2], XWF);
  gemm_wmma<<<2048, 256, 0, stream>>>(HCATR, IN1, WihP[3], 16, bih[3], bhh[3], XWB);
  lstm_scan2<<<2, 1024, LSTM_LDS_BYTES, stream>>>(XWF, XWB, WhhP[2], WhhP[3], mask,
                                                  H1CAT, 0, IN1, H1BREV, 0, HID);
  reverse_kernel<<<BL, 64, 0, stream>>>(H1BREV, HID, H1CAT, IN1, HID, HID, LEN);

  // ---- emissions + viterbi ----
  emis_kernel<<<BL, 32, 0, stream>>>(H1CAT, ffW, ffb, EM);
  viterbi_kernel<<<Bsz, 32, 0, stream>>>(EM, mask, trans, startv, endv, out, LEN);
}
